// Block_48137993453612
// MI455X (gfx1250) — matrix operand order repaired and checked by reading, not compiled, not run on hardware
//
#include <hip/hip_runtime.h>
#include <hip/hip_bf16.h>
#include <math.h>

typedef __attribute__((ext_vector_type(16))) __bf16 bf16x16;
typedef __attribute__((ext_vector_type(8)))  float  floatx8;

#define BM 64
#define BN 64
#define BKK 32

__device__ __forceinline__ float gelu_f(float x) {
    const float c = 0.7978845608028654f;
    return 0.5f * x * (1.0f + tanhf(c * (x + 0.044715f * x * x * x)));
}

// ---------------------------------------------------------------------------
// Generic batched GEMM: C[z] = act(A[z] @ B[z] + bias) [* scale, += if ACCUM]
// TA/TB: 0 = row-major view, 1 = transposed view (addr = k*ld + m / n*ld + k).
// fp32 inputs split into bf16 hi/lo while staging; LDS holds tiles in
// FRAGMENT ORDER: slot[tile][lane][0..15] contiguous 32B, so each WMMA operand
// is two aligned ds_load_b128 with no repacking.
// Staging: (1) per-thread int offsets precomputed once and advanced by a
// loop-invariant step (no 64-bit muls in-loop); (2) all 16 global loads issue
// into distinct registers first (one clause, one wait), then convert+store.
// Batch z: offA = (z/divA)*sA1 + (z%divA)*sA2; offB likewise unless `bsel`
// gives a device-side expert index: offB = bsel[z]*sB1.
// ---------------------------------------------------------------------------
template<int GELU, int ACCUM, int TA, int TB>
__global__ __launch_bounds__(256) void gemm_wmma_kernel(
    const float* __restrict__ A, int lda, int divA, long sA1, long sA2,
    const float* __restrict__ Bm, int ldb, int divB, long sB1, long sB2,
    const int*  __restrict__ bsel,
    const float* __restrict__ bias, long strideBias,
    float* __restrict__ C, int ldc, long strideC,
    const float* __restrict__ scale_ptr,
    int M, int Nc, int K)
{
    // fragment-order LDS: [4 tiles][32 lanes][16 elems]
    __shared__ __attribute__((aligned(32))) __bf16 AsH[4 * 32 * 16];
    __shared__ __attribute__((aligned(32))) __bf16 AsL[4 * 32 * 16];
    __shared__ __attribute__((aligned(32))) __bf16 BsH[4 * 32 * 16];
    __shared__ __attribute__((aligned(32))) __bf16 BsL[4 * 32 * 16];

    const int z = blockIdx.z;
    const long offA = (long)(z / divA) * sA1 + (long)(z % divA) * sA2;
    const int  eb   = bsel ? bsel[z] : 0;
    const long offB = bsel ? (long)eb * sB1
                           : ((long)(z / divB) * sB1 + (long)(z % divB) * sB2);
    const long offBias = bsel ? (long)eb * strideBias : 0;

    const int row0 = blockIdx.y * BM;
    const int col0 = blockIdx.x * BN;
    const int tid  = threadIdx.x;
    const int lane = tid & 31;
    const int wave = tid >> 5;
    const int tm   = wave & 3;          // wave's 16-row tile within 64
    const int tn0  = (wave >> 2) * 2;   // wave's first 16-col tile (owns 2)

    floatx8 acc0 = {}; floatx8 acc1 = {};

    const float* Ab = A + offA;
    const float* Bb = Bm + offB;

    // ---- precompute per-thread staging offsets (32-bit) + LDS slots ----
    int aoff[8], apos[8], boff[8], bpos[8];
    #pragma unroll
    for (int i = 0; i < 8; ++i) {
        int idx = i * 256 + tid;
        int m, k;
        if (TA) { m = idx & 63; k = idx >> 6; }     // m fastest: coalesced k*lda+m
        else    { m = idx >> 5; k = idx & 31; }     // k fastest: coalesced m*lda+k
        aoff[i] = TA ? (k * lda + row0 + m) : ((row0 + m) * lda + k);
        int grp    = k >> 3;
        int lane_s = (m & 15) + ((grp & 1) << 4);
        apos[i] = (((m >> 4) * 32 + lane_s) << 4) + (((grp >> 1) << 3) + (k & 7));

        int n, kk;
        if (TB) { kk = idx & 31; n = idx >> 5; }    // k fastest: coalesced n*ldb+k
        else    { n = idx & 63; kk = idx >> 6; }    // n fastest: coalesced k*ldb+n
        boff[i] = TB ? ((col0 + n) * ldb + kk) : (kk * ldb + col0 + n);
        int lane_b = (n & 15) + (((kk >> 3) & 1) << 4);
        bpos[i] = (((n >> 4) * 32 + lane_b) << 4) + (((kk >> 4) << 3) + (kk & 7));
    }
    const int aStep = TA ? BKK * lda : BKK;
    const int bStep = TB ? BKK : BKK * ldb;

    for (int k0 = 0; k0 < K; k0 += BKK) {
        // ---- phase 1: issue all 16 global loads (one clause, full MLP) ----
        float av[8], bv[8];
        #pragma unroll
        for (int i = 0; i < 8; ++i) { av[i] = Ab[aoff[i]]; aoff[i] += aStep; }
        #pragma unroll
        for (int i = 0; i < 8; ++i) { bv[i] = Bb[boff[i]]; boff[i] += bStep; }
        if (k0 + BKK < K) {  // gfx1250 global_prefetch_b8 of next A tile
            __builtin_prefetch(Ab + aoff[0], 0, 1);
        }
        // ---- phase 2: split fp32 -> bf16 hi/lo, store to fragment slots ----
        #pragma unroll
        for (int i = 0; i < 8; ++i) {
            __bf16 h = (__bf16)av[i];
            AsH[apos[i]] = h;
            AsL[apos[i]] = (__bf16)(av[i] - (float)h);
        }
        #pragma unroll
        for (int i = 0; i < 8; ++i) {
            __bf16 h = (__bf16)bv[i];
            BsH[bpos[i]] = h;
            BsL[bpos[i]] = (__bf16)(bv[i] - (float)h);
        }
        __syncthreads();

        // ---- aligned vector fragment loads (2x ds_load_b128 each) ----
        const int fa  = ((tm * 32 + lane) << 4);
        const int fb0 = ((tn0 * 32 + lane) << 4);
        const int fb1 = (((tn0 + 1) * 32 + lane) << 4);
        bf16x16 aH  = *(const bf16x16*)(AsH + fa);
        bf16x16 aL  = *(const bf16x16*)(AsL + fa);
        bf16x16 b0H = *(const bf16x16*)(BsH + fb0);
        bf16x16 b0L = *(const bf16x16*)(BsL + fb0);
        bf16x16 b1H = *(const bf16x16*)(BsH + fb1);
        bf16x16 b1L = *(const bf16x16*)(BsL + fb1);

        // fp32-via-bf16-split: lo terms first, then dominant hi*hi
        acc0 = __builtin_amdgcn_wmma_f32_16x16x32_bf16(false, aL, false, b0H, (short)0, acc0, false, false);
        acc0 = __builtin_amdgcn_wmma_f32_16x16x32_bf16(false, aH, false, b0L, (short)0, acc0, false, false);
        acc0 = __builtin_amdgcn_wmma_f32_16x16x32_bf16(false, aH, false, b0H, (short)0, acc0, false, false);
        acc1 = __builtin_amdgcn_wmma_f32_16x16x32_bf16(false, aL, false, b1H, (short)0, acc1, false, false);
        acc1 = __builtin_amdgcn_wmma_f32_16x16x32_bf16(false, aH, false, b1L, (short)0, acc1, false, false);
        acc1 = __builtin_amdgcn_wmma_f32_16x16x32_bf16(false, aH, false, b1H, (short)0, acc1, false, false);
        __syncthreads();
    }

    // ---- epilogue (f32 C layout: VGPR r -> M = r + (lane>>4)*8, N = lane&15)
    float* Cb = C + (long)z * strideC;
    const float scale = (ACCUM && scale_ptr) ? scale_ptr[z] : 1.0f;
    const int nl0 = (lane & 15) + tn0 * 16;
    const int cg0 = col0 + nl0, cg1 = cg0 + 16;
    const float bias0 = bias ? bias[offBias + cg0] : 0.0f;
    const float bias1 = bias ? bias[offBias + cg1] : 0.0f;
    #pragma unroll
    for (int r = 0; r < 8; ++r) {
        int mloc = tm * 16 + (lane >> 4) * 8 + r;
        int row  = row0 + mloc;
        float v0 = acc0[r] + bias0;
        float v1 = acc1[r] + bias1;
        if (GELU) { v0 = gelu_f(v0); v1 = gelu_f(v1); }
        int p0 = row * ldc + cg0;
        int p1 = row * ldc + cg1;
        if (ACCUM) { Cb[p0] += scale * v0; Cb[p1] += scale * v1; }
        else       { Cb[p0] = v0;          Cb[p1] = v1; }
    }
}

// ---------------------------------------------------------------------------
// softmax over rows of 64, scaled by 1/sqrt(N) beforehand
// ---------------------------------------------------------------------------
__global__ void softmax_rows_kernel(float* __restrict__ w, int cols, float scale) {
    __shared__ float red[64];
    long row = blockIdx.x;
    int  t   = threadIdx.x;
    float v = w[row * cols + t] * scale;
    red[t] = v; __syncthreads();
    for (int s = 32; s > 0; s >>= 1) { if (t < s) red[t] = fmaxf(red[t], red[t + s]); __syncthreads(); }
    float mx = red[0]; __syncthreads();
    float e = expf(v - mx);
    red[t] = e; __syncthreads();
    for (int s = 32; s > 0; s >>= 1) { if (t < s) red[t] += red[t + s]; __syncthreads(); }
    w[row * cols + t] = e / red[0];
}

// y = a + b; per-block partial (sum, sumsq) for global layernorm stats
__global__ void add_reduce_kernel(const float* __restrict__ a, const float* __restrict__ b,
                                  float* __restrict__ y, float* __restrict__ partials, long n) {
    float s = 0.f, s2 = 0.f;
    for (long i = (long)blockIdx.x * blockDim.x + threadIdx.x; i < n;
         i += (long)gridDim.x * blockDim.x) {
        float v = a[i] + b[i];
        y[i] = v; s += v; s2 += v * v;
    }
    __shared__ float r1[256], r2[256];
    r1[threadIdx.x] = s; r2[threadIdx.x] = s2; __syncthreads();
    for (int st = 128; st > 0; st >>= 1) {
        if (threadIdx.x < st) { r1[threadIdx.x] += r1[threadIdx.x + st]; r2[threadIdx.x] += r2[threadIdx.x + st]; }
        __syncthreads();
    }
    if (threadIdx.x == 0) { partials[blockIdx.x * 2] = r1[0]; partials[blockIdx.x * 2 + 1] = r2[0]; }
}

__global__ void finalize_stats_kernel(const float* __restrict__ partials,
                                      float* __restrict__ stats, int nblocks, long n) {
    __shared__ float r1[256], r2[256];
    float s = 0.f, s2 = 0.f;
    for (int i = threadIdx.x; i < nblocks; i += 256) { s += partials[i * 2]; s2 += partials[i * 2 + 1]; }
    r1[threadIdx.x] = s; r2[threadIdx.x] = s2; __syncthreads();
    for (int st = 128; st > 0; st >>= 1) {
        if (threadIdx.x < st) { r1[threadIdx.x] += r1[threadIdx.x + st]; r2[threadIdx.x] += r2[threadIdx.x + st]; }
        __syncthreads();
    }
    if (threadIdx.x == 0) {
        float mean = r1[0] / (float)n;
        float var  = (r2[0] - (float)n * mean * mean) / (float)(n - 1);  // ddof=1
        stats[0] = mean;
        stats[1] = rsqrtf(var + 1e-12f);
    }
}

__global__ void layernorm_apply_kernel(const float* __restrict__ y, const float* __restrict__ stats,
                                       const float* __restrict__ w, const float* __restrict__ b,
                                       float* __restrict__ out, long n, int Dd) {
    float mean = stats[0], rstd = stats[1];
    for (long i = (long)blockIdx.x * blockDim.x + threadIdx.x; i < n;
         i += (long)gridDim.x * blockDim.x) {
        int d = (int)(i % Dd);
        out[i] = (y[i] - mean) * rstd * w[d] + b[d];
    }
}

__global__ void colmean_kernel(const float* __restrict__ x, float* __restrict__ xbar,
                               int Nrows, int Dcols) {
    int bd = blockIdx.x;
    int b  = bd / Dcols, d = bd % Dcols;
    const float* p = x + (long)b * Nrows * Dcols + d;
    float s = 0.f;
    for (int n = threadIdx.x; n < Nrows; n += blockDim.x) s += p[(long)n * Dcols];
    __shared__ float red[256];
    red[threadIdx.x] = s; __syncthreads();
    for (int st = 128; st > 0; st >>= 1) {
        if (threadIdx.x < st) red[threadIdx.x] += red[threadIdx.x + st];
        __syncthreads();
    }
    if (threadIdx.x == 0) xbar[bd] = red[0] / (float)Nrows;
}

// gate = softmax(xbar @ r_w + r_b), top-2 per batch (device-side routing)
__global__ void gate_topk_kernel(const float* __restrict__ xbar, const float* __restrict__ rw,
                                 const float* __restrict__ rb, float* __restrict__ gvals,
                                 int* __restrict__ gsel, int D, int E, int Bt) {
    __shared__ float logits[16];
    int t = threadIdx.x;
    if (t < Bt * E) {
        int b = t / E, e = t % E;
        float s = rb[e];
        for (int d = 0; d < D; ++d) s += xbar[b * D + d] * rw[(long)d * E + e];
        logits[t] = s;
    }
    __syncthreads();
    if (t < Bt) {
        int b = t;
        float mx = -1e30f;
        for (int e = 0; e < E; ++e) mx = fmaxf(mx, logits[b * E + e]);
        float den = 0.f, p[8];
        for (int e = 0; e < E; ++e) { p[e] = expf(logits[b * E + e] - mx); den += p[e]; }
        for (int e = 0; e < E; ++e) p[e] /= den;
        int i1 = 0;
        for (int e = 1; e < E; ++e) if (p[e] > p[i1]) i1 = e;
        int i2 = (i1 == 0) ? 1 : 0;
        for (int e = 0; e < E; ++e) if (e != i1 && p[e] > p[i2]) i2 = e;
        gvals[b * 2 + 0] = p[i1]; gvals[b * 2 + 1] = p[i2];
        gsel[b * 2 + 0] = i1;     gsel[b * 2 + 1] = i2;
    }
}

__global__ void fill_zero_kernel(float* __restrict__ p, long n) {
    for (long i = (long)blockIdx.x * blockDim.x + threadIdx.x; i < n;
         i += (long)gridDim.x * blockDim.x) p[i] = 0.f;
}

// ---------------------------------------------------------------------------
extern "C" void kernel_launch(void* const* d_in, const int* in_sizes, int n_in,
                              void* d_out, int out_size, void* d_ws, size_t ws_size,
                              hipStream_t stream) {
    const float* x    = (const float*)d_in[0];
    const float* W1w  = (const float*)d_in[1];
    const float* W1b  = (const float*)d_in[2];
    const float* W2w  = (const float*)d_in[3];
    const float* W2b  = (const float*)d_in[4];
    const float* rw   = (const float*)d_in[5];
    const float* rb   = (const float*)d_in[6];
    const float* fcw  = (const float*)d_in[7];
    const float* fcb  = (const float*)d_in[8];
    const float* pjw  = (const float*)d_in[9];
    const float* pjb  = (const float*)d_in[10];
    const float* ln1w = (const float*)d_in[11];
    const float* ln1b = (const float*)d_in[12];
    const float* ln2w = (const float*)d_in[13];
    const float* ln2b = (const float*)d_in[14];

    const int Bt = 2, Nn = 2048, D = 768, E = 8, H = 3072, NH = 12;
    const long tok   = (long)Bt * Nn;   // 4096
    const long total = tok * D;         // 3145728
    const int  RED_BLOCKS = 512;

    float* fws   = (float*)d_ws;
    float* qkv   = fws;                         // [4096, 2304]
    float* hmid  = qkv;                         // alias: [2048,3072] (qkv dead by then)
    float* swbuf = qkv + tok * 3 * D;           // [24, 64, 64]
    float* tmp   = swbuf + 24 * 4096;           // [B, 768, 2048]
    float* xres  = tmp + total;                 // attn_out -> y1 -> x1 (in place)
    float* macc  = xres + total;                // MoE accumulation -> y2
    float* parts = macc + total;                // 1024 partials
    float* stats = parts + 2 * RED_BLOCKS;      // mean, rstd
    float* xbar  = stats + 2;                   // [B, D]
    float* gvals = xbar + (long)Bt * D;         // [4]
    int*   gsel  = (int*)(gvals + 4);           // [4]

    // 1) qkv = x @ W1_w + W1_b      [4096,768]x[768,2304]
    gemm_wmma_kernel<0,0,0,0><<<dim3(3 * D / BN, tok / BM, 1), 256, 0, stream>>>(
        x, D, 1, 0, 0,
        W1w, 3 * D, 1, 0, 0,
        nullptr, W1b, 0,
        qkv, 3 * D, 0, nullptr,
        (int)tok, 3 * D, D);

    // 2) scores[b,h] = q @ k : q is [dh,N] column-view of Q, k is [N,dh] row-view of K
    gemm_wmma_kernel<0,0,1,0><<<dim3(1, 1, Bt * NH), 256, 0, stream>>>(
        qkv,     3 * D, NH, (long)Nn * 3 * D, 64,
        qkv + D, 3 * D, NH, (long)Nn * 3 * D, 64,
        nullptr, nullptr, 0,
        swbuf, 64, 4096, nullptr,
        64, 64, Nn);

    // 3) softmax(scores / sqrt(N)) over last axis (rows of 64)
    softmax_rows_kernel<<<Bt * NH * 64, 64, 0, stream>>>(swbuf, 64, 1.0f / sqrtf((float)Nn));

    // 4) tmp[b,h] = w @ v : [64,64]x[64,2048], v is [dh,N] column-view of V
    gemm_wmma_kernel<0,0,0,1><<<dim3(Nn / BN, 1, Bt * NH), 256, 0, stream>>>(
        swbuf, 64, 1, 4096, 0,
        qkv + 2 * D, 3 * D, NH, (long)Nn * 3 * D, 64,
        nullptr, nullptr, 0,
        tmp, Nn, (long)64 * Nn, nullptr,
        64, Nn, 64);

    // 5) attn = reshape(tmp) @ W2_w + W2_b  (raw row-major reinterpretation)
    gemm_wmma_kernel<0,0,0,0><<<dim3(D / BN, tok / BM, 1), 256, 0, stream>>>(
        tmp, D, 1, 0, 0,
        W2w, D, 1, 0, 0,
        nullptr, W2b, 0,
        xres, D, 0, nullptr,
        (int)tok, D, D);

    // 6-8) y1 = attn + x; global-scalar layernorm (unbiased var) -> x1 in place
    add_reduce_kernel<<<RED_BLOCKS, 256, 0, stream>>>(xres, x, xres, parts, total);
    finalize_stats_kernel<<<1, 256, 0, stream>>>(parts, stats, RED_BLOCKS, total);
    layernorm_apply_kernel<<<RED_BLOCKS, 256, 0, stream>>>(xres, stats, ln1w, ln1b, xres, total, D);

    // 9-10) gate = softmax((mean_n x1) @ r_w + r_b); top-2 routing on device
    colmean_kernel<<<Bt * D, 256, 0, stream>>>(xres, xbar, Nn, D);
    gate_topk_kernel<<<1, 32, 0, stream>>>(xbar, rw, rb, gvals, gsel, D, E, Bt);

    // 11) MoE accumulator = 0
    fill_zero_kernel<<<RED_BLOCKS, 256, 0, stream>>>(macc, total);

    // 12) only the top-2 experts per batch (identical numerics to dense+take)
    for (int s = 0; s < Bt * 2; ++s) {
        int b = s >> 1;
        // hmid = gelu(x1[b] @ fc_w[e] + fc_b[e]),  e = gsel[s] read on device
        gemm_wmma_kernel<1,0,0,0><<<dim3(H / BN, Nn / BM, 1), 256, 0, stream>>>(
            xres + (long)b * Nn * D, D, 1, 0, 0,
            fcw, H, 1, (long)D * H, 0,
            gsel + s, fcb, H,
            hmid, H, 0, nullptr,
            Nn, H, D);
        // macc[b] += gvals[s] * (hmid @ proj_w[e] + proj_b[e])
        gemm_wmma_kernel<0,1,0,0><<<dim3(D / BN, Nn / BM, 1), 256, 0, stream>>>(
            hmid, H, 1, 0, 0,
            pjw, D, 1, (long)H * D, 0,
            gsel + s, pjb, D,
            macc + (long)b * Nn * D, D, 0, gvals + s,
            Nn, D, H);
    }

    // 13-15) y2 = moe + x1; global layernorm -> d_out
    add_reduce_kernel<<<RED_BLOCKS, 256, 0, stream>>>(macc, xres, macc, parts, total);
    finalize_stats_kernel<<<1, 256, 0, stream>>>(parts, stats, RED_BLOCKS, total);
    layernorm_apply_kernel<<<RED_BLOCKS, 256, 0, stream>>>(macc, stats, ln2w, ln2b,
                                                           (float*)d_out, total, D);
}
